// SparseMoeBlock_75342316306793
// MI455X (gfx1250) — matrix operand order repaired and checked by reading, not compiled, not run on hardware
//
#include <hip/hip_runtime.h>
#include <hip/hip_bf16.h>
#include <math.h>

// Problem constants (match reference)
#define TOKENS 8192
#define HD     2048
#define NE     8

typedef __attribute__((ext_vector_type(16))) __bf16 v16bf;
typedef __attribute__((ext_vector_type(8)))  float  v8f;

// Async global->LDS DMA (CDNA5), guarded: fall back to uint4 copies if the
// toolchain doesn't declare the builtins.
#if defined(__has_builtin)
#  if __has_builtin(__builtin_amdgcn_global_load_async_to_lds_b128) && \
      __has_builtin(__builtin_amdgcn_s_wait_asynccnt)
#    define MOE_ASYNC_LDS 1
#  endif
#endif
#ifndef MOE_ASYNC_LDS
#  define MOE_ASYNC_LDS 0
#endif

#if MOE_ASYNC_LDS
// Parameter types per hipcc diagnostic: int4 (gcc vector_size) pointers,
// global operand in AS1 (printed "__device__"), LDS operand in AS3.
typedef int moe_v4i __attribute__((__vector_size__(4 * sizeof(int))));
typedef __attribute__((address_space(1))) moe_v4i moe_gv4i;
typedef __attribute__((address_space(3))) moe_v4i moe_lv4i;
__device__ inline void async_cp16(const __bf16* g, __bf16* l)
{
    __builtin_amdgcn_global_load_async_to_lds_b128((moe_gv4i*)g, (moe_lv4i*)l, 0, 0);
}
#endif

// ---------------------------------------------------------------------------
// Kernel 1: router. One wave per token. f32 exact logits -> top2 -> weights.
// ---------------------------------------------------------------------------
__global__ __launch_bounds__(256) void moe_router_kernel(
    const float* __restrict__ x, const float* __restrict__ gw,
    float* __restrict__ logits_out, int* __restrict__ topi,
    float* __restrict__ topw, int* __restrict__ counts)
{
    const int wave = threadIdx.x >> 5;
    const int lane = threadIdx.x & 31;
    const int t = blockIdx.x * 8 + wave;

    const float* xr = x + (size_t)t * HD;
    float acc[NE];
#pragma unroll
    for (int e = 0; e < NE; ++e) acc[e] = 0.0f;

    for (int d = lane; d < HD; d += 32) {
        const float xv = xr[d];
#pragma unroll
        for (int e = 0; e < NE; ++e) acc[e] += xv * gw[e * HD + d];
    }
#pragma unroll
    for (int e = 0; e < NE; ++e) {
        float v = acc[e];
#pragma unroll
        for (int o = 16; o > 0; o >>= 1) v += __shfl_xor(v, o, 32);
        acc[e] = v;
    }
    if (lane == 0) {
#pragma unroll
        for (int e = 0; e < NE; ++e) logits_out[(size_t)t * NE + e] = acc[e];
        int i1 = 0; float m1 = acc[0];
#pragma unroll
        for (int e = 1; e < NE; ++e) if (acc[e] > m1) { m1 = acc[e]; i1 = e; }
        int i2 = (i1 == 0) ? 1 : 0; float m2 = acc[i2];
#pragma unroll
        for (int e = 0; e < NE; ++e)
            if (e != i1 && acc[e] > m2) { m2 = acc[e]; i2 = e; }
        const float w2 = 1.0f / (1.0f + expf(m1 - m2));
        const float w1 = 1.0f - w2;
        topi[t * 2 + 0] = i1; topi[t * 2 + 1] = i2;
        topw[t * 2 + 0] = w1; topw[t * 2 + 1] = w2;
        atomicAdd(&counts[i1], 1);
        atomicAdd(&counts[i2], 1);
    }
}

// ---------------------------------------------------------------------------
// Kernel 2: exclusive prefix sum over 8 counts
// ---------------------------------------------------------------------------
__global__ void moe_offsets_kernel(const int* __restrict__ counts,
                                   int* __restrict__ offsets,
                                   int* __restrict__ cursor)
{
    if (threadIdx.x == 0) {
        int s = 0;
        for (int e = 0; e < NE; ++e) { offsets[e] = s; cursor[e] = s; s += counts[e]; }
        offsets[NE] = s;
    }
}

// ---------------------------------------------------------------------------
// Kernel 3: build per-expert token lists
// ---------------------------------------------------------------------------
__global__ __launch_bounds__(256) void moe_scatter_kernel(
    const int* __restrict__ topi, const float* __restrict__ topw,
    int* __restrict__ cursor, int* __restrict__ perm, float* __restrict__ pwt)
{
    const int t = blockIdx.x * 256 + threadIdx.x;
    if (t >= TOKENS) return;
#pragma unroll
    for (int k = 0; k < 2; ++k) {
        const int e = topi[t * 2 + k];
        const int pos = atomicAdd(&cursor[e], 1);
        perm[pos] = t;
        pwt[pos]  = topw[t * 2 + k];
    }
}

// ---------------------------------------------------------------------------
// Shared helpers for the WMMA GEMMs
// ---------------------------------------------------------------------------
#define ASTR 40                 // padded LDS row stride (elements): conflict-free
#define ABKS (64  * ASTR)       // A buffer elements
#define BBKS (128 * ASTR)       // B buffer elements (pre path, BN=128)
#define BKS  (64  * ASTR)       // fallback buffers (BN=64)

__device__ inline void cvt8_store(const float4 a, const float4 b,
                                  __bf16* __restrict__ hi, __bf16* __restrict__ lo)
{
    const float f[8] = { a.x, a.y, a.z, a.w, b.x, b.y, b.z, b.w };
    __bf16 h[8], l[8];
#pragma unroll
    for (int i = 0; i < 8; ++i) {
        const __bf16 t = (__bf16)f[i];
        h[i] = t;
        l[i] = (__bf16)(f[i] - (float)t);
    }
    __builtin_memcpy(hi, h, 16);
    __builtin_memcpy(lo, l, 16);
}

// A operand (16x32, MxK): lanes 0-15 = rows; elems 0..7 = K h*8..+7,
// elems 8..15 = K h*8+16..+23  (h = lane>>4)
__device__ inline v16bf load_a(const __bf16* base, int lane)
{
    const int m = lane & 15, half = lane >> 4;
    const __bf16* p = base + m * ASTR + half * 8;
    v16bf r;
    __builtin_memcpy(&r, p, 16);
    __builtin_memcpy(((__bf16*)&r) + 8, p + 16, 16);
    return r;
}

// B operand (32x16, KxN): lane n = col (lane&15); elems 0..7 = K kh*8..+7,
// elems 8..15 = K kh*8+16..+23
__device__ inline v16bf load_b(const __bf16* base, int lane)
{
    const int n = lane & 15, kh = lane >> 4;
    const __bf16* p = base + n * ASTR + kh * 8;
    v16bf r;
    __builtin_memcpy(&r, p, 16);
    __builtin_memcpy(((__bf16*)&r) + 8, p + 16, 16);
    return r;
}

__device__ inline v8f wmma3(v16bf ah, v16bf al, v16bf bh, v16bf bl, v8f c)
{
    // x*w ~= xh*wh + xh*wl + xl*wh   (drops xl*wl ~ 2^-16 relative)
    c = __builtin_amdgcn_wmma_f32_16x16x32_bf16(false, ah, false, bh, (short)0, c, false, false);
    c = __builtin_amdgcn_wmma_f32_16x16x32_bf16(false, ah, false, bl, (short)0, c, false, false);
    c = __builtin_amdgcn_wmma_f32_16x16x32_bf16(false, al, false, bh, (short)0, c, false, false);
    return c;
}

// ---------------------------------------------------------------------------
// One-shot f32 -> (bf16 hi, bf16 lo) split, 8 elements / thread, b128 in/out.
// ---------------------------------------------------------------------------
__global__ __launch_bounds__(256) void cvt_hilo_kernel(
    const float* __restrict__ src, __bf16* __restrict__ hi,
    __bf16* __restrict__ lo, int nchunk8)
{
    const int i = blockIdx.x * 256 + threadIdx.x;
    if (i >= nchunk8) return;
    const float4 a = ((const float4*)src)[2 * i];
    const float4 b = ((const float4*)src)[2 * i + 1];
    cvt8_store(a, b, hi + (size_t)i * 8, lo + (size_t)i * 8);
}

// ---------------------------------------------------------------------------
// Kernel 4a (main path): grouped GEMM from pre-split bf16 hi/lo arrays.
//   256 threads = 8 wave32 (4m x 2n), 64x128 tile, K-step 32, double-buffered
//   LDS. Staging via async global->LDS DMA when available.
// ---------------------------------------------------------------------------
__global__ __launch_bounds__(256) void moe_gemm_pre_kernel(
    const __bf16* __restrict__ Xh, const __bf16* __restrict__ Xl,
    const __bf16* __restrict__ Wh, const __bf16* __restrict__ Wl,
    const int* __restrict__ perm, const float* __restrict__ pwt,
    const int* __restrict__ offsets, float* __restrict__ out)
{
    __shared__ __bf16 sAh[2 * ABKS], sAl[2 * ABKS];
    __shared__ __bf16 sBh[2 * BBKS], sBl[2 * BBKS];
    __shared__ int   s_tok[64];
    __shared__ float s_w[64];

    const int e    = blockIdx.z;
    const int mt   = blockIdx.y;
    const int hblk = blockIdx.x * 128;

    const int obeg = offsets[e];
    const int cnt  = offsets[e + 1] - obeg;
    if (mt * 64 >= cnt) return;          // uniform early-exit

    const int tid = threadIdx.x;
    if (tid < 64) {
        const int loc  = mt * 64 + tid;
        const int locc = (loc < cnt) ? loc : (cnt - 1);
        s_tok[tid] = perm[obeg + locc];
        s_w[tid]   = (loc < cnt) ? pwt[obeg + locc] : 0.0f;
    }
    __syncthreads();

    // staging: thread -> (row sr, 8-wide chunk kb); B also handles row sr+64
    const int sr = tid >> 2;
    const int kb = (tid & 3) * 8;
    const __bf16* axh = Xh + (size_t)s_tok[sr] * HD + kb;
    const __bf16* axl = Xl + (size_t)s_tok[sr] * HD + kb;
    const size_t wb0 = ((size_t)e * HD + hblk + sr) * HD + kb;
    const __bf16* bwh0 = Wh + wb0;
    const __bf16* bwl0 = Wl + wb0;
    const __bf16* bwh1 = bwh0 + (size_t)64 * HD;
    const __bf16* bwl1 = bwl0 + (size_t)64 * HD;
    const int siA  = sr * ASTR + kb;
    const int siB1 = (sr + 64) * ASTR + kb;

    // stage 0 into buffer 0
#if MOE_ASYNC_LDS
    async_cp16(axh,  sAh + siA);
    async_cp16(axl,  sAl + siA);
    async_cp16(bwh0, sBh + siA);
    async_cp16(bwl0, sBl + siA);
    async_cp16(bwh1, sBh + siB1);
    async_cp16(bwl1, sBl + siB1);
    __builtin_amdgcn_s_wait_asynccnt(0);
#else
    *(uint4*)(sAh + siA)  = *(const uint4*)(axh);
    *(uint4*)(sAl + siA)  = *(const uint4*)(axl);
    *(uint4*)(sBh + siA)  = *(const uint4*)(bwh0);
    *(uint4*)(sBl + siA)  = *(const uint4*)(bwl0);
    *(uint4*)(sBh + siB1) = *(const uint4*)(bwh1);
    *(uint4*)(sBl + siB1) = *(const uint4*)(bwl1);
#endif
    __syncthreads();

    const int lane = tid & 31;
    const int wid  = tid >> 5;
    const int wm   = wid >> 1;           // 0..3 -> rows wm*16..+16
    const int wn   = wid & 1;            // 0..1 -> cols wn*64..+64
    v8f acc[4] = { {}, {}, {}, {} };

    const int NKI = HD / 32;
#pragma unroll 2
    for (int kt = 0; kt < NKI; ++kt) {
        const int bufA = (kt & 1) * ABKS;
        const int bufB = (kt & 1) * BBKS;
        const bool more = (kt + 1) < NKI;
#if MOE_ASYNC_LDS
        if (more) {                      // DMA next stage; overlaps the WMMAs
            const int d0 = (kt + 1) * 32;
            const int nA = ((kt + 1) & 1) * ABKS;
            const int nB = ((kt + 1) & 1) * BBKS;
            async_cp16(axh  + d0, sAh + nA + siA);
            async_cp16(axl  + d0, sAl + nA + siA);
            async_cp16(bwh0 + d0, sBh + nB + siA);
            async_cp16(bwl0 + d0, sBl + nB + siA);
            async_cp16(bwh1 + d0, sBh + nB + siB1);
            async_cp16(bwl1 + d0, sBl + nB + siB1);
        }
#else
        uint4 ra_h, ra_l, rb0_h, rb0_l, rb1_h, rb1_l;
        if (more) {                      // prefetch next stage to registers
            const int d0 = (kt + 1) * 32;
            ra_h  = *(const uint4*)(axh  + d0);
            ra_l  = *(const uint4*)(axl  + d0);
            rb0_h = *(const uint4*)(bwh0 + d0);
            rb0_l = *(const uint4*)(bwl0 + d0);
            rb1_h = *(const uint4*)(bwh1 + d0);
            rb1_l = *(const uint4*)(bwl1 + d0);
        }
#endif
        {
            const __bf16* Ahb = sAh + bufA + (wm * 16) * ASTR;
            const __bf16* Alb = sAl + bufA + (wm * 16) * ASTR;
            const __bf16* Bhb = sBh + bufB + (wn * 64) * ASTR;
            const __bf16* Blb = sBl + bufB + (wn * 64) * ASTR;
            const v16bf ah = load_a(Ahb, lane);
            const v16bf al = load_a(Alb, lane);
#pragma unroll
            for (int t = 0; t < 4; ++t) {
                const v16bf bh = load_b(Bhb + (16 * t) * ASTR, lane);
                const v16bf bl = load_b(Blb + (16 * t) * ASTR, lane);
                acc[t] = wmma3(ah, al, bh, bl, acc[t]);
            }
        }
#if MOE_ASYNC_LDS
        if (more) __builtin_amdgcn_s_wait_asynccnt(0);
#else
        if (more) {
            const int nA = ((kt + 1) & 1) * ABKS;
            const int nB = ((kt + 1) & 1) * BBKS;
            *(uint4*)(sAh + nA + siA)  = ra_h;
            *(uint4*)(sAl + nA + siA)  = ra_l;
            *(uint4*)(sBh + nB + siA)  = rb0_h;
            *(uint4*)(sBl + nB + siA)  = rb0_l;
            *(uint4*)(sBh + nB + siB1) = rb1_h;
            *(uint4*)(sBl + nB + siB1) = rb1_l;
        }
#endif
        __syncthreads();
    }

    // epilogue: elem r of v8f: M = r + 8*(lane>>4), N = lane&15
    const int mhalf = (lane >> 4) * 8;
    const int n     = lane & 15;
#pragma unroll
    for (int r = 0; r < 8; ++r) {
        const int   ml  = wm * 16 + mhalf + r;
        const int   tok = s_tok[ml];
        const float w   = s_w[ml];
        float* op = out + (size_t)tok * HD + hblk + wn * 64 + n;
#pragma unroll
        for (int t = 0; t < 4; ++t)
            atomicAdd(op + 16 * t, acc[t][r] * w);
    }
}

// ---------------------------------------------------------------------------
// Kernel 4b (fallback if workspace too small): 64x64 tile, convert in-loop.
// ---------------------------------------------------------------------------
__global__ __launch_bounds__(256) void moe_gemm_fb_kernel(
    const float* __restrict__ x, const float* __restrict__ ew,
    const int* __restrict__ perm, const float* __restrict__ pwt,
    const int* __restrict__ offsets, float* __restrict__ out)
{
    __shared__ __bf16 sAh[2 * BKS], sAl[2 * BKS], sBh[2 * BKS], sBl[2 * BKS];
    __shared__ int   s_tok[64];
    __shared__ float s_w[64];

    const int e    = blockIdx.z;
    const int mt   = blockIdx.y;
    const int hblk = blockIdx.x * 64;

    const int obeg = offsets[e];
    const int cnt  = offsets[e + 1] - obeg;
    if (mt * 64 >= cnt) return;

    const int tid = threadIdx.x;
    if (tid < 64) {
        const int loc  = mt * 64 + tid;
        const int locc = (loc < cnt) ? loc : (cnt - 1);
        s_tok[tid] = perm[obeg + locc];
        s_w[tid]   = (loc < cnt) ? pwt[obeg + locc] : 0.0f;
    }
    __syncthreads();

    const int sr = tid >> 2;
    const int kb = (tid & 3) * 8;
    const float* aptr = x  + (size_t)s_tok[sr] * HD + kb;
    const float* bptr = ew + (size_t)e * HD * HD + (size_t)(hblk + sr) * HD + kb;
    const int sidx = sr * ASTR + kb;

    {
        const float4 a0 = *(const float4*)(aptr);
        const float4 a1 = *(const float4*)(aptr + 4);
        const float4 b0 = *(const float4*)(bptr);
        const float4 b1 = *(const float4*)(bptr + 4);
        cvt8_store(a0, a1, sAh + sidx, sAl + sidx);
        cvt8_store(b0, b1, sBh + sidx, sBl + sidx);
    }
    __syncthreads();

    const int lane = tid & 31;
    const int wid  = tid >> 5;
    const int wm   = wid >> 1;
    const int wn   = wid & 1;
    v8f acc0 = {};
    v8f acc1 = {};

    const int NKI = HD / 32;
#pragma unroll 2
    for (int kt = 0; kt < NKI; ++kt) {
        const int buf  = (kt & 1) * BKS;
        const bool more = (kt + 1) < NKI;
        float4 a0, a1, b0, b1;
        if (more) {
            const int d0 = (kt + 1) * 32;
            a0 = *(const float4*)(aptr + d0);
            a1 = *(const float4*)(aptr + d0 + 4);
            b0 = *(const float4*)(bptr + d0);
            b1 = *(const float4*)(bptr + d0 + 4);
        }
        {
            const __bf16* Ahb = sAh + buf + (wm * 16) * ASTR;
            const __bf16* Alb = sAl + buf + (wm * 16) * ASTR;
            const __bf16* Bhb = sBh + buf + (wn * 32) * ASTR;
            const __bf16* Blb = sBl + buf + (wn * 32) * ASTR;
            const v16bf ah  = load_a(Ahb, lane);
            const v16bf al  = load_a(Alb, lane);
            const v16bf bh0 = load_b(Bhb, lane);
            const v16bf bl0 = load_b(Blb, lane);
            const v16bf bh1 = load_b(Bhb + 16 * ASTR, lane);
            const v16bf bl1 = load_b(Blb + 16 * ASTR, lane);
            acc0 = wmma3(ah, al, bh0, bl0, acc0);
            acc1 = wmma3(ah, al, bh1, bl1, acc1);
        }
        if (more) {
            const int nb = ((kt + 1) & 1) * BKS;
            cvt8_store(a0, a1, sAh + nb + sidx, sAl + nb + sidx);
            cvt8_store(b0, b1, sBh + nb + sidx, sBl + nb + sidx);
        }
        __syncthreads();
    }

    const int mhalf = (lane >> 4) * 8;
    const int n     = lane & 15;
#pragma unroll
    for (int r = 0; r < 8; ++r) {
        const int   ml  = wm * 16 + mhalf + r;
        const int   tok = s_tok[ml];
        const float w   = s_w[ml];
        float* op = out + (size_t)tok * HD + hblk + wn * 32 + n;
        atomicAdd(op,      acc0[r] * w);
        atomicAdd(op + 16, acc1[r] * w);
    }
}

// ---------------------------------------------------------------------------
// Host launch.
// Output layout: [T*HD] final_hidden_states, then [T*NE] router_logits.
// ---------------------------------------------------------------------------
extern "C" void kernel_launch(void* const* d_in, const int* in_sizes, int n_in,
                              void* d_out, int out_size, void* d_ws, size_t ws_size,
                              hipStream_t stream)
{
    const float* x  = (const float*)d_in[0];
    const float* gw = (const float*)d_in[1];
    const float* ew = (const float*)d_in[2];

    float* out    = (float*)d_out;
    float* logits = out + (size_t)TOKENS * HD;

    char* ws      = (char*)d_ws;
    int*   counts  = (int*)(ws);
    int*   cursor  = (int*)(ws + 32);
    int*   offsets = (int*)(ws + 64);
    int*   topi    = (int*)(ws + 128);
    float* topw    = (float*)(ws + 128 + (size_t)TOKENS * 2 * 4);
    int*   perm    = (int*)(ws + 128 + (size_t)TOKENS * 4 * 4);
    float* pwt     = (float*)(ws + 128 + (size_t)TOKENS * 6 * 4);

    const size_t XBYTES = (size_t)TOKENS * HD * 2;        // 32 MiB per part
    const size_t WBYTES = (size_t)NE * HD * HD * 2;       // 64 MiB per part
    const size_t BIG0   = (size_t)1 << 20;
    const size_t need   = BIG0 + 2 * XBYTES + 2 * WBYTES; // ~193 MiB

    (void)hipMemsetAsync(counts, 0, 64, stream);
    (void)hipMemsetAsync(out, 0, (size_t)TOKENS * HD * sizeof(float), stream);

    moe_router_kernel<<<TOKENS / 8, 256, 0, stream>>>(x, gw, logits, topi, topw, counts);
    moe_offsets_kernel<<<1, 32, 0, stream>>>(counts, offsets, cursor);
    moe_scatter_kernel<<<TOKENS / 256, 256, 0, stream>>>(topi, topw, cursor, perm, pwt);

    if (ws_size >= need) {
        __bf16* Xh = (__bf16*)(ws + BIG0);
        __bf16* Xl = (__bf16*)(ws + BIG0 + XBYTES);
        __bf16* Wh = (__bf16*)(ws + BIG0 + 2 * XBYTES);
        __bf16* Wl = (__bf16*)(ws + BIG0 + 2 * XBYTES + WBYTES);

        const int nx8 = (TOKENS * HD) / 8;
        const int nw8 = (NE * HD * HD) / 8;
        cvt_hilo_kernel<<<nx8 / 256, 256, 0, stream>>>(x,  Xh, Xl, nx8);
        cvt_hilo_kernel<<<nw8 / 256, 256, 0, stream>>>(ew, Wh, Wl, nw8);

        dim3 grid(HD / 128, TOKENS / 64, NE);
        moe_gemm_pre_kernel<<<grid, 256, 0, stream>>>(Xh, Xl, Wh, Wl,
                                                      perm, pwt, offsets, out);
    } else {
        dim3 grid(HD / 64, TOKENS / 64, NE);
        moe_gemm_fb_kernel<<<grid, 256, 0, stream>>>(x, ew, perm, pwt, offsets, out);
    }
}
